// GraphSAGEFraudDetector_26096221290642
// MI455X (gfx1250) — compile-verified, matrix operand order checked by reading, not run on hardware
//
#include <hip/hip_runtime.h>

// ---------------------------------------------------------------------------
// GraphSAGE (3-layer, mean aggr) for MI455X / gfx1250.
// Strategy: transform-then-aggregate (mean is linear), WMMA bf16 GEMMs,
// L2-resident scatter via global f32 atomics (tables << 192MB L2).
// ---------------------------------------------------------------------------

typedef __attribute__((ext_vector_type(16))) __bf16 v16bf;
typedef __attribute__((ext_vector_type(8)))  float  v8f;

#define N_NODES 100000
#define N_EDGES 1600000

__device__ __forceinline__ unsigned short f2bf(float f) {
    union { float f; unsigned u; } c; c.f = f;
    unsigned u = c.u;
    unsigned r = (u + 0x7FFFu + ((u >> 16) & 1u)) >> 16;   // round-to-nearest-even
    return (unsigned short)r;
}
__device__ __forceinline__ float bf2f(unsigned short h) {
    union { unsigned u; float f; } c; c.u = ((unsigned)h) << 16;
    return c.f;
}

union FragB16 { v16bf v; uint4 q[2]; };

// --------------------------- prep kernels ----------------------------------

__global__ void zero_f32(float* __restrict__ p, size_t n) {
    size_t i = (size_t)blockIdx.x * blockDim.x + threadIdx.x;
    size_t s = (size_t)gridDim.x * blockDim.x;
    for (; i < n; i += s) p[i] = 0.0f;
}

__global__ void f32_to_bf16(const float* __restrict__ src,
                            unsigned short* __restrict__ dst, size_t n) {
    size_t i = (size_t)blockIdx.x * blockDim.x + threadIdx.x;
    size_t s = (size_t)gridDim.x * blockDim.x;
    for (; i < n; i += s) dst[i] = f2bf(src[i]);
}

// Pack W[K x 64] (row-major f32) into WMMA B-fragment order:
// frag = ks*4 + nt; within frag: lane l (0..31), element e (0..15):
//   n = nt*16 + (l&15), half = l>>4, j = e>>1, bit = e&1
//   k = ks*32 + (j>=4 ? 16 : 0) + 8*half + 2*(j&3) + bit
__global__ void pack_weight_frags(const float* __restrict__ W,
                                  unsigned short* __restrict__ P, int K) {
    int total = K * 64;
    for (int t = blockIdx.x * blockDim.x + threadIdx.x; t < total;
         t += gridDim.x * blockDim.x) {
        int e    = t & 15;
        int l    = (t >> 4) & 31;
        int frag = t >> 9;
        int nt   = frag & 3;
        int ks   = frag >> 2;
        int n    = nt * 16 + (l & 15);
        int half = l >> 4;
        int j    = e >> 1, bit = e & 1;
        int k    = ks * 32 + ((j >= 4) ? 16 : 0) + 8 * half + 2 * (j & 3) + bit;
        P[t] = f2bf(W[k * 64 + n]);
    }
}

// --------------------------- WMMA GEMM -------------------------------------
// yl = act @ Wl ; yr = act @ Wr + bias.  act: [N,K] bf16, N multiple of 16.
// One wave per 16-node tile, full 64 output features (4 N-tiles), both
// weight matrices (A-fragment reuse).  8 x v_wmma_f32_16x16x32_bf16 per K-step.
__global__ __launch_bounds__(256)
void sage_gemm_wmma(const unsigned short* __restrict__ act,
                    const unsigned short* __restrict__ wlp,
                    const unsigned short* __restrict__ wrp,
                    const float* __restrict__ bias,
                    float* __restrict__ yl, float* __restrict__ yr,
                    int nTiles, int K) {
    int lane = threadIdx.x & 31;
    int wave = threadIdx.x >> 5;
    int tile = blockIdx.x * 8 + wave;
    if (tile >= nTiles) return;                 // wave-uniform: EXEC stays all-1s

    int m = lane & 15, half = lane >> 4;
    const unsigned short* arow = act + (size_t)(tile * 16 + m) * K;
    int kSteps = K >> 5;

    v8f accL[4] = {};
    v8f accR[4] = {};

    for (int ks = 0; ks < kSteps; ++ks) {
        // A fragment: two contiguous 16B loads per lane (ISA 16-bit A layout)
        FragB16 a;
        a.q[0] = *(const uint4*)(arow + ks * 32 + 8 * half);
        a.q[1] = *(const uint4*)(arow + ks * 32 + 16 + 8 * half);
#pragma unroll
        for (int nt = 0; nt < 4; ++nt) {
            FragB16 bl, br;
            const uint4* blp =
                (const uint4*)(wlp + ((size_t)((ks * 4 + nt) * 32 + lane)) * 16);
            const uint4* brp =
                (const uint4*)(wrp + ((size_t)((ks * 4 + nt) * 32 + lane)) * 16);
            bl.q[0] = blp[0]; bl.q[1] = blp[1];
            br.q[0] = brp[0]; br.q[1] = brp[1];
            accL[nt] = __builtin_amdgcn_wmma_f32_16x16x32_bf16(
                false, a.v, false, bl.v, (short)0, accL[nt], false, false);
            accR[nt] = __builtin_amdgcn_wmma_f32_16x16x32_bf16(
                false, a.v, false, br.v, (short)0, accR[nt], false, false);
        }
    }

    // C/D layout: lane -> N = lane&15, VGPR v -> M = v + 8*(lane>>4)
    int n0 = lane & 15;
#pragma unroll
    for (int nt = 0; nt < 4; ++nt) {
        float bv = bias[nt * 16 + n0];
#pragma unroll
        for (int v = 0; v < 8; ++v) {
            int row    = tile * 16 + v + 8 * half;
            size_t idx = (size_t)row * 64 + nt * 16 + n0;
            yl[idx] = accL[nt][v];
            yr[idx] = accR[nt][v] + bv;
        }
    }
}

// --------------------------- edge kernels ----------------------------------

__global__ void degree_kernel(const int* __restrict__ dst,
                              float* __restrict__ deg, int E) {
    int e = blockIdx.x * blockDim.x + threadIdx.x;
    if (e < E) atomicAdd(&deg[dst[e]], 1.0f);
}

// agg[dst] += yl[src]  (64 features per edge; yl/agg tables are L2-resident)
__global__ void scatter_add64(const int* __restrict__ src,
                              const int* __restrict__ dst,
                              const float* __restrict__ yl,
                              float* __restrict__ agg, int E) {
    int e = blockIdx.x * blockDim.x + threadIdx.x;
    if (e >= E) return;
    const float4* row = (const float4*)(yl + (size_t)src[e] * 64);
    float* out = agg + (size_t)dst[e] * 64;
#pragma unroll
    for (int i = 0; i < 16; ++i) {
        float4 v = row[i];
        atomicAdd(out + 4 * i + 0, v.x);
        atomicAdd(out + 4 * i + 1, v.y);
        atomicAdd(out + 4 * i + 2, v.z);
        atomicAdd(out + 4 * i + 3, v.w);
    }
}

// h = relu(agg/max(deg,1) + yr) as bf16; re-zero agg for the next layer.
__global__ void combine_relu(const float* __restrict__ yr,
                             float* __restrict__ agg,
                             const float* __restrict__ deg,
                             unsigned short* __restrict__ h, int n64) {
    int gid = blockIdx.x * blockDim.x + threadIdx.x;
    if (gid >= n64) return;
    int i   = gid >> 6;
    float a = agg[gid];
    agg[gid] = 0.0f;
    float v = a / fmaxf(deg[i], 1.0f) + yr[gid];
    h[gid]  = f2bf(fmaxf(v, 0.0f));
}

// Layer 2 (out dim 1): per-node scalars sl = h.wl2, sr = h.wr2 + b2;
// also zero the scalar aggregation buffer.
__global__ void final_dots(const unsigned short* __restrict__ h,
                           const float* __restrict__ wl2,
                           const float* __restrict__ wr2,
                           const float* __restrict__ b2,
                           float* __restrict__ sl, float* __restrict__ sr,
                           float* __restrict__ aggS, int N) {
    int i = blockIdx.x * blockDim.x + threadIdx.x;
    if (i >= N) return;
    const unsigned short* row = h + (size_t)i * 64;
    float a = 0.0f, b = 0.0f;
#pragma unroll
    for (int f = 0; f < 64; ++f) {
        float hv = bf2f(row[f]);
        a += hv * wl2[f];
        b += hv * wr2[f];
    }
    sl[i] = a;
    sr[i] = b + b2[0];
    aggS[i] = 0.0f;
}

__global__ void scatter_add1(const int* __restrict__ src,
                             const int* __restrict__ dst,
                             const float* __restrict__ sl,
                             float* __restrict__ aggS, int E) {
    int e = blockIdx.x * blockDim.x + threadIdx.x;
    if (e < E) atomicAdd(&aggS[dst[e]], sl[src[e]]);
}

__global__ void sigmoid_out(const float* __restrict__ aggS,
                            const float* __restrict__ sr,
                            const float* __restrict__ deg,
                            float* __restrict__ out, int N) {
    int i = blockIdx.x * blockDim.x + threadIdx.x;
    if (i >= N) return;
    float v = aggS[i] / fmaxf(deg[i], 1.0f) + sr[i];
    out[i] = 1.0f / (1.0f + expf(-v));
}

// --------------------------- launcher --------------------------------------

extern "C" void kernel_launch(void* const* d_in, const int* in_sizes, int n_in,
                              void* d_out, int out_size, void* d_ws, size_t ws_size,
                              hipStream_t stream) {
    const float* x   = (const float*)d_in[0];
    const int*   ei  = (const int*)d_in[1];        // edge_index [2, E]
    const float* wl0 = (const float*)d_in[2];
    const float* wr0 = (const float*)d_in[3];
    const float* b0  = (const float*)d_in[4];
    const float* wl1 = (const float*)d_in[5];
    const float* wr1 = (const float*)d_in[6];
    const float* b1  = (const float*)d_in[7];
    const float* wl2 = (const float*)d_in[8];
    const float* wr2 = (const float*)d_in[9];
    const float* b2  = (const float*)d_in[10];
    float* out = (float*)d_out;

    const int N = N_NODES, E = N_EDGES;
    const int* src = ei;
    const int* dst = ei + E;

    // Workspace carve-up (256B aligned). agg aliases xb: xb is dead after the
    // layer-0 GEMM, and stream ordering serializes the reuse.
    char* ws = (char*)d_ws;
    size_t off = 0;
    auto carve = [&](size_t bytes) -> char* {
        char* p = ws + off;
        off += (bytes + 255) & ~(size_t)255;
        return p;
    };
    unsigned short* xb = (unsigned short*)carve((size_t)N * 128 * 2); // 25.6 MB
    float* agg = (float*)xb;                                          // alias
    float* yl  = (float*)carve((size_t)N * 64 * 4);                   // 25.6 MB
    float* yr  = (float*)carve((size_t)N * 64 * 4);                   // 25.6 MB
    unsigned short* h = (unsigned short*)carve((size_t)N * 64 * 2);   // 12.8 MB
    float* deg  = (float*)carve((size_t)N * 4);
    float* sl   = (float*)carve((size_t)N * 4);
    float* sr   = (float*)carve((size_t)N * 4);
    float* aggS = (float*)carve((size_t)N * 4);
    unsigned short* wl0p = (unsigned short*)carve(128 * 64 * 2);
    unsigned short* wr0p = (unsigned short*)carve(128 * 64 * 2);
    unsigned short* wl1p = (unsigned short*)carve(64 * 64 * 2);
    unsigned short* wr1p = (unsigned short*)carve(64 * 64 * 2);

    const int nTiles = N / 16;                 // 6250 (exact)
    const int gemmBlocks = (nTiles + 7) / 8;   // 8 waves / block
    const int edgeBlocks = (E + 255) / 256;

    // Prep: bf16 activations, fragment-packed bf16 weights, degrees.
    f32_to_bf16<<<2048, 256, 0, stream>>>(x, xb, (size_t)N * 128);
    pack_weight_frags<<<32, 256, 0, stream>>>(wl0, wl0p, 128);
    pack_weight_frags<<<32, 256, 0, stream>>>(wr0, wr0p, 128);
    pack_weight_frags<<<16, 256, 0, stream>>>(wl1, wl1p, 64);
    pack_weight_frags<<<16, 256, 0, stream>>>(wr1, wr1p, 64);
    zero_f32<<<512, 256, 0, stream>>>(deg, (size_t)N);
    degree_kernel<<<edgeBlocks, 256, 0, stream>>>(dst, deg, E);

    // Layer 0: GEMMs first (reads xb), then agg (aliases xb) is zeroed+used.
    sage_gemm_wmma<<<gemmBlocks, 256, 0, stream>>>(xb, wl0p, wr0p, b0, yl, yr,
                                                   nTiles, 128);
    zero_f32<<<4096, 256, 0, stream>>>(agg, (size_t)N * 64);
    scatter_add64<<<edgeBlocks, 256, 0, stream>>>(src, dst, yl, agg, E);
    combine_relu<<<(N * 64 + 255) / 256, 256, 0, stream>>>(yr, agg, deg, h,
                                                           N * 64);

    // Layer 1 (agg was re-zeroed by combine_relu).
    sage_gemm_wmma<<<gemmBlocks, 256, 0, stream>>>(h, wl1p, wr1p, b1, yl, yr,
                                                   nTiles, 64);
    scatter_add64<<<edgeBlocks, 256, 0, stream>>>(src, dst, yl, agg, E);
    combine_relu<<<(N * 64 + 255) / 256, 256, 0, stream>>>(yr, agg, deg, h,
                                                           N * 64);

    // Layer 2: scalar transform-then-aggregate, then sigmoid.
    final_dots<<<(N + 255) / 256, 256, 0, stream>>>(h, wl2, wr2, b2, sl, sr,
                                                    aggS, N);
    scatter_add1<<<edgeBlocks, 256, 0, stream>>>(src, dst, sl, aggS, E);
    sigmoid_out<<<(N + 255) / 256, 256, 0, stream>>>(aggS, sr, deg, out, N);
}
